// MemKDMClassModel_84121229460122
// MI455X (gfx1250) — matrix operand order repaired and checked by reading, not compiled, not run on hardware
//
#include <hip/hip_runtime.h>
#include <hip/hip_bf16.h>

typedef __attribute__((ext_vector_type(16))) __bf16 v16bf;
typedef __attribute__((ext_vector_type(8)))  float  v8f;

namespace {
constexpr int D = 128;   // embedding dim
constexpr int C = 100;   // classes
constexpr int K = 64;    // neighbors per query
constexpr float EPS = 1e-12f;
}

// One workgroup (256 threads = 8 wave32) per query row b.
// Phase 1: coalesced float4 gather of 64 neighbor rows + wave reduction -> d2
// Phase 2: w = exp(-d2/sigma^2), normalized (wave 0)
// Phase 3: probs = w @ Y^2 via v_wmma_f32_16x16x32_bf16 (broadcast-A trick),
//          with batched, unconditional gathers (clamp+mask instead of branches)
__global__ __launch_bounds__(256) void kdm_probs_kernel(
    const float* __restrict__ x_enc,     // (B, 128)
    const int*   __restrict__ neighbors, // (B, 64)
    const float* __restrict__ samples_x, // (1e6, 128)
    const float* __restrict__ samples_y, // (1e6, 100)
    const float* __restrict__ sigma,     // (1,)
    float*       __restrict__ out)       // (B, 100)
{
    __shared__ float4 s_x4[D / 4];   // query row, float4-aligned
    __shared__ float  s_d2[K];
    __shared__ float  s_wn[K];
    __shared__ int    s_nbr[K];

    const int b    = blockIdx.x;
    const int tid  = threadIdx.x;
    const int lane = tid & 31;
    const int wave = tid >> 5;

    // ---- stage query row + neighbor ids into LDS ----
    if (tid < D) {
        reinterpret_cast<float*>(s_x4)[tid] = x_enc[(size_t)b * D + tid];
    } else if (tid < D + K) {
        s_nbr[tid - D] = neighbors[(size_t)b * K + (tid - D)];
    }
    __syncthreads();

    // ---- phase 1: squared distances; wave w owns neighbors [8w, 8w+8) ----
    const float4 xq = s_x4[lane];            // 4 contiguous dims per lane
    #pragma unroll
    for (int j = 0; j < 8; ++j) {
        const int k   = wave * 8 + j;
        const int nbr = s_nbr[k];
        const float4 xn =
            reinterpret_cast<const float4*>(samples_x + (size_t)nbr * D)[lane];
        const float e0 = xq.x - xn.x, e1 = xq.y - xn.y;
        const float e2 = xq.z - xn.z, e3 = xq.w - xn.w;
        float s = e0 * e0 + e1 * e1 + e2 * e2 + e3 * e3;
        #pragma unroll
        for (int m = 16; m >= 1; m >>= 1) s += __shfl_xor(s, m, 32);
        if (lane == 0) s_d2[k] = s;
    }
    __syncthreads();

    // ---- phase 2: RBF weights + normalization (wave 0 only) ----
    if (wave == 0) {
        const float sg    = sigma[0];
        const float invs2 = 1.0f / (sg * sg);      // kvals^2 = exp(-d2/sigma^2)
        const float w0 = __expf(-s_d2[lane]      * invs2);
        const float w1 = __expf(-s_d2[lane + 32] * invs2);
        float ssum = w0 + w1;
        #pragma unroll
        for (int m = 16; m >= 1; m >>= 1) ssum += __shfl_xor(ssum, m, 32);
        const float inv = 1.0f / (ssum + EPS);
        s_wn[lane]      = w0 * inv;
        s_wn[lane + 32] = w1 * inv;
    }
    __syncthreads();

    // ---- phase 3: probs[c0..c0+15] = sum_k w[k] * y[nbr_k][c]^2 via WMMA ----
    // A(16x32) = 16 identical rows of the w k-chunk  (A layout:
    //   lanes 0-15 hold K 0-15; lanes 16-31 hold K 16-31)
    // B(32x16) = Y^2 tile (lanes 0-15: K 0-15 of col N=lane;
    //   lanes 16-31: K 16-31 of col N=lane-16) -> every row of D is the answer.
    if (wave < 7) {
        const int  c0    = wave * 16;
        const int  col   = c0 + (lane & 15);
        const bool colOK = (col < C);
        const int  colC  = colOK ? col : (C - 1);   // clamp: loads unconditional
        const int  kRowBase = (lane < 16) ? 0 : 16;
        const int  aSel     = (lane < 16) ? 0 : 16;

        v8f acc = {};
        #pragma unroll
        for (int kb = 0; kb < K; kb += 32) {
            // pass 1: batched LDS reads of the 16 neighbor row ids
            int rows[16];
            #pragma unroll
            for (int e = 0; e < 16; ++e)
                rows[e] = s_nbr[kb + kRowBase + e];

            // pass 2: batched, unconditional global gathers (64B-contiguous
            // per half-wave) -> one load clause, one wait
            float yraw[16];
            #pragma unroll
            for (int e = 0; e < 16; ++e)
                yraw[e] = samples_y[(size_t)rows[e] * C + colC];

            // pass 3: pack operands
            const int aBase = kb + aSel;
            v16bf amat, bmat;
            #pragma unroll
            for (int e = 0; e < 16; ++e) {
                amat[e] = (__bf16)s_wn[aBase + e];
                const float yv = colOK ? yraw[e] : 0.0f;
                bmat[e] = (__bf16)(yv * yv);
            }
            acc = __builtin_amdgcn_wmma_f32_16x16x32_bf16(
                false, amat, false, bmat, (short)0, acc, false, false);
        }
        if (lane < 16 && colOK)
            out[(size_t)b * C + col] = acc[0];       // D row M=0, N=lane
    }
}

extern "C" void kernel_launch(void* const* d_in, const int* in_sizes, int n_in,
                              void* d_out, int out_size, void* d_ws, size_t ws_size,
                              hipStream_t stream) {
    (void)n_in; (void)d_ws; (void)ws_size; (void)out_size;
    const float* x_enc     = (const float*)d_in[0];
    const int*   neighbors = (const int*)  d_in[1];
    const float* samples_x = (const float*)d_in[2];
    const float* samples_y = (const float*)d_in[3];
    const float* sigma     = (const float*)d_in[4];
    float* out = (float*)d_out;

    const int B = in_sizes[0] / D;   // 4096
    kdm_probs_kernel<<<B, 256, 0, stream>>>(x_enc, neighbors, samples_x,
                                            samples_y, sigma, out);
}
